// DeeProBot_MoE_52518860095672
// MI455X (gfx1250) — matrix operand order repaired and checked by reading, not compiled, not run on hardware
//
#include <hip/hip_runtime.h>
#include <hip/hip_bf16.h>

// ---------------- problem constants ----------------
#define B_    32768
#define IN_   512
#define E_    8
#define H_    128
#define MO_   256
#define OUT_  2
#define TILE  64                    // rows (slots) per expert-kernel block
#define NSLOT (2*B_ + E_*TILE)      // 66048: 2B slots + per-expert pad to TILE
#define GATE_BLOCKS (B_/8)          // 4096 gating blocks (8 rows each)

// WMMA vector types (probe-confirmed signatures)
typedef __attribute__((ext_vector_type(16))) __bf16   bf16x16;
typedef __attribute__((ext_vector_type(8)))  float    f32x8;
typedef __attribute__((ext_vector_type(4)))  unsigned u32x4;
typedef __attribute__((ext_vector_type(2)))  unsigned u32x2;
union FragBF { unsigned u[8]; u32x4 q[2]; bf16x16 v; };

static __device__ __forceinline__ unsigned short f2bf(float f) {
    unsigned u = __float_as_uint(f);
    u += 0x7FFFu + ((u >> 16) & 1u);        // round-to-nearest-even
    return (unsigned short)(u >> 16);
}
static __device__ __forceinline__ unsigned pack2bf(float a, float b) {
    return (unsigned)f2bf(a) | ((unsigned)f2bf(b) << 16);
}

// ---------------- K0a: pack W1 [E,512,128] f32 -> bf16 B-fragment order ----------------
// flat = (((e*8 + nt)*16 + ks)*32 + lane)*8 + p ; VGPR p holds K = ks*32 + (lane>=16)*16 + 2p,2p+1 ; N = nt*16 + lane%16
__global__ __launch_bounds__(256) void pack_w1_kernel(const float* __restrict__ W1, unsigned* __restrict__ w1pk) {
    int i = blockIdx.x * 256 + threadIdx.x;
    if (i >= E_*IN_*H_/2) return;
    int p = i & 7, lane = (i >> 3) & 31, ks = (i >> 8) & 15, nt = (i >> 12) & 7, e = i >> 15;
    int n = nt*16 + (lane & 15);
    int k = ks*32 + ((lane >> 4) & 1)*16 + p*2;
    w1pk[i] = pack2bf(W1[(size_t)(e*IN_ + k)*H_ + n], W1[(size_t)(e*IN_ + k + 1)*H_ + n]);
}

// ---------------- K0b: pack W2 [E,128,256] ----------------
__global__ __launch_bounds__(256) void pack_w2_kernel(const float* __restrict__ W2, unsigned* __restrict__ w2pk) {
    int i = blockIdx.x * 256 + threadIdx.x;
    if (i >= E_*H_*MO_/2) return;
    int p = i & 7, lane = (i >> 3) & 31, ks = (i >> 8) & 3, nt = (i >> 10) & 15, e = i >> 14;
    int n = nt*16 + (lane & 15);
    int k = ks*32 + ((lane >> 4) & 1)*16 + p*2;
    w2pk[i] = pack2bf(W2[(size_t)(e*H_ + k)*MO_ + n], W2[(size_t)(e*H_ + k + 1)*MO_ + n]);
}

// ---------------- K1: zero bucket arrays (padding slots -> row 0, gate 0) ----------------
__global__ __launch_bounds__(256) void init_kernel(int* __restrict__ brow, float* __restrict__ bgate) {
    int i = blockIdx.x * 256 + threadIdx.x;
    if (i < NSLOT) { brow[i] = 0; bgate[i] = 0.0f; }
}

// ---------------- K2: gating: logits, top-2 softmax, per-block partials ----------------
__global__ __launch_bounds__(256) void gate_kernel(const float* __restrict__ x, const float* __restrict__ wg,
                                                   int* __restrict__ top_idx, float* __restrict__ gate_val,
                                                   float* __restrict__ imp_part, int* __restrict__ cnt_part) {
    __shared__ float s_imp[E_];
    __shared__ int   s_cnt[E_];
    int tid = threadIdx.x, lane = tid & 31, w = tid >> 5;
    if (tid < E_) { s_imp[tid] = 0.0f; s_cnt[tid] = 0; }
    __syncthreads();
    int b = blockIdx.x * 8 + w;
    float acc[E_];
    #pragma unroll
    for (int e = 0; e < E_; ++e) acc[e] = 0.0f;
    const float* xr = x + (size_t)b * IN_;
    for (int i = lane; i < IN_; i += 32) {
        float xv = xr[i];
        const float* wr = wg + i * E_;
        #pragma unroll
        for (int e = 0; e < E_; ++e) acc[e] = fmaf(xv, wr[e], acc[e]);
    }
    #pragma unroll
    for (int e = 0; e < E_; ++e)
        #pragma unroll
        for (int off = 16; off > 0; off >>= 1) acc[e] += __shfl_xor(acc[e], off, 32);
    if (lane == 0) {
        int i0 = 0; float v0 = acc[0];
        #pragma unroll
        for (int e = 1; e < E_; ++e) if (acc[e] > v0) { v0 = acc[e]; i0 = e; }
        int i1 = -1; float v1 = -3.4e38f;
        #pragma unroll
        for (int e = 0; e < E_; ++e) if (e != i0 && acc[e] > v1) { v1 = acc[e]; i1 = e; }
        float g1e = expf(v1 - v0);                 // stable softmax over top-2 (v0 >= v1)
        float inv = 1.0f / (1.0f + g1e);
        float g0 = inv, g1 = g1e * inv;
        top_idx[b*2] = i0;  top_idx[b*2+1] = i1;
        gate_val[b*2] = g0; gate_val[b*2+1] = g1;
        atomicAdd(&s_imp[i0], g0); atomicAdd(&s_imp[i1], g1);
        atomicAdd(&s_cnt[i0], 1);  atomicAdd(&s_cnt[i1], 1);
    }
    __syncthreads();
    if (tid < E_) { imp_part[blockIdx.x*E_ + tid] = s_imp[tid]; cnt_part[blockIdx.x*E_ + tid] = s_cnt[tid]; }
}

// ---------------- K3: reduce partials -> padded offsets + cv^2 loss ----------------
__global__ __launch_bounds__(256) void reduce_kernel(const float* __restrict__ imp_part, const int* __restrict__ cnt_part,
                                                     int* __restrict__ offs, int* __restrict__ counts2,
                                                     float* __restrict__ d_loss) {
    __shared__ float s_imp[E_];
    __shared__ int   s_cnt[E_];
    int tid = threadIdx.x, lane = tid & 31, e = tid >> 5;   // one wave per expert
    float si = 0.0f; int sc = 0;
    for (int g = lane; g < GATE_BLOCKS; g += 32) { si += imp_part[g*E_ + e]; sc += cnt_part[g*E_ + e]; }
    #pragma unroll
    for (int off = 16; off > 0; off >>= 1) { si += __shfl_xor(si, off, 32); sc += __shfl_xor(sc, off, 32); }
    if (lane == 0) { s_imp[e] = si; s_cnt[e] = sc; }
    __syncthreads();
    if (tid == 0) {
        int off = 0;
        for (int k = 0; k < E_; ++k) { offs[k] = off; off += (s_cnt[k] + TILE - 1) & ~(TILE - 1); counts2[k] = 0; }
        offs[E_] = off;
        float mi = 0.0f, ml = 0.0f;
        for (int k = 0; k < E_; ++k) { mi += s_imp[k]; ml += (float)s_cnt[k]; }
        mi *= (1.0f/E_); ml *= (1.0f/E_);
        float vi = 0.0f, vl = 0.0f;
        for (int k = 0; k < E_; ++k) {
            float d = s_imp[k] - mi;         vi += d * d;
            float d2 = (float)s_cnt[k] - ml; vl += d2 * d2;
        }
        vi *= (1.0f/(E_-1)); vl *= (1.0f/(E_-1));   // ddof=1
        *d_loss = (vi / (mi*mi + 1e-10f) + vl / (ml*ml + 1e-10f)) * 1e-2f;
    }
}

// ---------------- K4: block-aggregated scatter into expert buckets ----------------
__global__ __launch_bounds__(256) void scatter_kernel(const int* __restrict__ top_idx, const float* __restrict__ gate_val,
                                                      const int* __restrict__ offs, int* __restrict__ counts2,
                                                      int* __restrict__ brow, float* __restrict__ bgate,
                                                      int* __restrict__ slot_of) {
    __shared__ int s_cnt[E_], s_base[E_];
    int tid = threadIdx.x;
    if (tid < E_) s_cnt[tid] = 0;
    __syncthreads();
    int b = blockIdx.x * 256 + tid;                 // grid is exactly B_/256
    int e0 = top_idx[b*2], e1 = top_idx[b*2+1];
    int r0 = atomicAdd(&s_cnt[e0], 1);
    int r1 = atomicAdd(&s_cnt[e1], 1);
    __syncthreads();
    if (tid < E_) s_base[tid] = atomicAdd(&counts2[tid], s_cnt[tid]);
    __syncthreads();
    int p0 = offs[e0] + s_base[e0] + r0;
    int p1 = offs[e1] + s_base[e1] + r1;
    brow[p0] = b; bgate[p0] = gate_val[b*2];   slot_of[b*2]   = p0;
    brow[p1] = b; bgate[p1] = gate_val[b*2+1]; slot_of[b*2+1] = p1;
}

// ---------------- K5: routed expert kernel (bf16 WMMA) ----------------
// 64 gathered rows per block, all of one expert. 8 waves:
//   GEMM1: wave w owns N-tile w of h[64,128]   (16 k-steps of 32 over K=512)
//   GEMM2: wave w owns N-tiles {2w,2w+1} of t[64,256] (4 k-steps over K=128)
// LDS row strides are 16B-multiples (ds_load_b128-able) that rotate rows by
// 4 banks, so the 32-lane fragment read touches every bank exactly twice.
#define XS 520    // x tile row stride (ushort): 1040 B = 65*16, 260 dwords (+4 banks/row)
#define HS 136    // h tile row stride (ushort):  272 B = 17*16,  68 dwords (+4 banks/row)
#define TS 260    // t tile row stride (float): +8-row groups land 32 banks apart
#define SMEM_BYTES (TILE*TS*4 + TILE*HS*2 + 256 + 256)   // xs(=66560) aliases ts(=66560)

__global__ __launch_bounds__(256) void expert_kernel(const float* __restrict__ x,
                                                     const unsigned* __restrict__ w1pk,
                                                     const unsigned* __restrict__ w2pk,
                                                     const float* __restrict__ b1,
                                                     const float* __restrict__ b2,
                                                     const float* __restrict__ wout,
                                                     const int* __restrict__ offs,
                                                     const int* __restrict__ brow,
                                                     const float* __restrict__ bgate,
                                                     float* __restrict__ contrib) {
    extern __shared__ unsigned char smem[];
    unsigned short* xs = (unsigned short*)smem;                      // [64][XS] bf16 (dead after GEMM1)
    float*          ts = (float*)smem;                               // [64][TS] f32 (aliases xs)
    unsigned short* hs = (unsigned short*)(smem + TILE*TS*4);        // [64][HS] bf16
    int*          rIdx = (int*)(smem + TILE*TS*4 + TILE*HS*2);       // [64]
    float*       rGate = (float*)(smem + TILE*TS*4 + TILE*HS*2 + 256);

    const int tileStart = blockIdx.x * TILE;
    if (tileStart >= offs[E_]) return;                               // past last padded slot
    int e = 0;
    while (e < E_-1 && tileStart >= offs[e+1]) ++e;

    const int tid  = threadIdx.x;
    const int lane = tid & 31;
    const int w    = tid >> 5;
    const int arow = lane & 15;
    const int m8   = (lane >> 4) << 3;      // 2*m8 = K-offset (A frag), m8 = M-offset (C frag) for upper lane half

    if (tid < TILE) { rIdx[tid] = brow[tileStart + tid]; rGate[tid] = bgate[tileStart + tid]; }
    __syncthreads();

    // ---- stage x rows -> LDS bf16 (float4 loads, 8B packed LDS stores) ----
    for (int i = tid; i < TILE*(IN_/4); i += 256) {
        int r = i >> 7, c4 = i & 127;                                // 128 float4 per row
        float4 f = ((const float4*)(x + (size_t)rIdx[r] * IN_))[c4];
        u32x2 d;
        d.x = pack2bf(f.x, f.y);
        d.y = pack2bf(f.z, f.w);
        *(u32x2*)(xs + r*XS + c4*4) = d;                             // 8B aligned
    }
    __syncthreads();

    // ---- GEMM1: h[64,128] = relu(xs @ W1[e] + b1[e]) ----
    f32x8 acc1[4];
    #pragma unroll
    for (int mt = 0; mt < 4; ++mt) acc1[mt] = (f32x8)(0.0f);
    const unsigned* w1b = w1pk + (((size_t)(e*8 + w)*16)*32)*8;
    for (int ks = 0; ks < 16; ++ks) {
        const unsigned* bp = w1b + ks*256 + lane*8;
        __builtin_prefetch(bp + 256, 0, 1);          // next k-step's B fragment
        FragBF bF;
        bF.q[0] = ((const u32x4*)bp)[0];
        bF.q[1] = ((const u32x4*)bp)[1];
        #pragma unroll
        for (int mt = 0; mt < 4; ++mt) {
            FragBF aF;
            const unsigned short* xr = xs + (mt*16 + arow)*XS + ks*32 + 2*m8;
            aF.q[0] = *(const u32x4*)(xr);           // K pairs p0..p3 (16B aligned)
            aF.q[1] = *(const u32x4*)(xr + 8);       // K pairs p4..p7 (+16B)
            acc1[mt] = __builtin_amdgcn_wmma_f32_16x16x32_bf16(false, aF.v, false, bF.v,
                                                               (short)0, acc1[mt], false, false);
        }
    }
    {   // C layout: VGPR j -> M = j (+8 for lanes 16..31), N = lane%16
        int col = w*16 + arow;
        float bb = b1[e*H_ + col];
        #pragma unroll
        for (int mt = 0; mt < 4; ++mt)
            #pragma unroll
            for (int j = 0; j < 8; ++j) {
                float v = acc1[mt][j] + bb;
                hs[(mt*16 + j + m8)*HS + col] = f2bf(v > 0.0f ? v : 0.0f);
            }
    }
    __syncthreads();   // hs complete; xs now dead for everyone

    // ---- GEMM2: t[64,256] = hs @ W2[e] + b2[e] ----
    f32x8 acc2[4][2];
    #pragma unroll
    for (int mt = 0; mt < 4; ++mt) { acc2[mt][0] = (f32x8)(0.0f); acc2[mt][1] = (f32x8)(0.0f); }
    for (int ks = 0; ks < 4; ++ks) {
        const unsigned* bp0 = w2pk + (((size_t)(e*16 + 2*w    )*4 + ks)*32 + lane)*8;
        const unsigned* bp1 = w2pk + (((size_t)(e*16 + 2*w + 1)*4 + ks)*32 + lane)*8;
        __builtin_prefetch(bp0 + 256, 0, 1);
        FragBF bF0, bF1;
        bF0.q[0] = ((const u32x4*)bp0)[0];
        bF0.q[1] = ((const u32x4*)bp0)[1];
        bF1.q[0] = ((const u32x4*)bp1)[0];
        bF1.q[1] = ((const u32x4*)bp1)[1];
        #pragma unroll
        for (int mt = 0; mt < 4; ++mt) {
            FragBF aF;
            const unsigned short* hr = hs + (mt*16 + arow)*HS + ks*32 + 2*m8;
            aF.q[0] = *(const u32x4*)(hr);
            aF.q[1] = *(const u32x4*)(hr + 8);
            acc2[mt][0] = __builtin_amdgcn_wmma_f32_16x16x32_bf16(false, aF.v, false, bF0.v,
                                                                  (short)0, acc2[mt][0], false, false);
            acc2[mt][1] = __builtin_amdgcn_wmma_f32_16x16x32_bf16(false, aF.v, false, bF1.v,
                                                                  (short)0, acc2[mt][1], false, false);
        }
    }
    #pragma unroll
    for (int nti = 0; nti < 2; ++nti) {
        int col = (2*w + nti)*16 + arow;
        float bb = b2[e*MO_ + col];
        #pragma unroll
        for (int mt = 0; mt < 4; ++mt)
            #pragma unroll
            for (int j = 0; j < 8; ++j)
                ts[(mt*16 + j + m8)*TS + col] = acc2[mt][nti][j] + bb;
    }
    __syncthreads();

    // ---- fused softmax(256) + Wout head: contrib = gate * softmax(t) @ Wout ----
    for (int rr = 0; rr < 8; ++rr) {
        int r = w*8 + rr;
        const float* trow = ts + r*TS;
        float v[8], m = -3.4e38f;
        #pragma unroll
        for (int i = 0; i < 8; ++i) { v[i] = trow[lane + 32*i]; m = fmaxf(m, v[i]); }
        #pragma unroll
        for (int off = 16; off > 0; off >>= 1) m = fmaxf(m, __shfl_xor(m, off, 32));
        float s = 0.0f, c0 = 0.0f, c1 = 0.0f;
        #pragma unroll
        for (int i = 0; i < 8; ++i) {
            float p = __expf(v[i] - m);
            int col = lane + 32*i;
            s += p;
            c0 = fmaf(p, wout[col*2 + 0], c0);
            c1 = fmaf(p, wout[col*2 + 1], c1);
        }
        #pragma unroll
        for (int off = 16; off > 0; off >>= 1) {
            s  += __shfl_xor(s, off, 32);
            c0 += __shfl_xor(c0, off, 32);
            c1 += __shfl_xor(c1, off, 32);
        }
        if (lane == 0) {
            float g = rGate[r] / s;     // (softmax @ Wout) = (exp @ Wout)/sum
            contrib[(size_t)(tileStart + r)*2 + 0] = g * c0;
            contrib[(size_t)(tileStart + r)*2 + 1] = g * c1;
        }
    }
}

// ---------------- K6: deterministic gather-combine -> out ----------------
__global__ __launch_bounds__(256) void combine_kernel(const float* __restrict__ contrib, const int* __restrict__ slot_of,
                                                      const float* __restrict__ bout, float* __restrict__ out) {
    int b = blockIdx.x * 256 + threadIdx.x;          // grid exactly B_/256
    int s0 = slot_of[b*2], s1 = slot_of[b*2+1];
    out[b*2 + 0] = bout[0] + contrib[(size_t)s0*2 + 0] + contrib[(size_t)s1*2 + 0];
    out[b*2 + 1] = bout[1] + contrib[(size_t)s0*2 + 1] + contrib[(size_t)s1*2 + 1];
}

// ---------------- launch ----------------
extern "C" void kernel_launch(void* const* d_in, const int* in_sizes, int n_in,
                              void* d_out, int out_size, void* d_ws, size_t ws_size,
                              hipStream_t stream) {
    (void)in_sizes; (void)n_in; (void)out_size; (void)ws_size;
    const float* x    = (const float*)d_in[0];
    // d_in[1] = cat_prop: unused by the reference
    const float* wg   = (const float*)d_in[2];
    const float* W1   = (const float*)d_in[3];
    const float* b1   = (const float*)d_in[4];
    const float* W2   = (const float*)d_in[5];
    const float* b2   = (const float*)d_in[6];
    const float* Wout = (const float*)d_in[7];
    const float* bout = (const float*)d_in[8];
    float* out = (float*)d_out;

    unsigned char* ws = (unsigned char*)d_ws;
    size_t o = 0;
    auto carve = [&](size_t bytes) -> void* {
        void* p = ws + o;
        o = (o + bytes + 255) & ~(size_t)255;
        return p;
    };
    int*      top_idx  = (int*)     carve((size_t)2*B_*4);
    float*    gate_val = (float*)   carve((size_t)2*B_*4);
    float*    imp_part = (float*)   carve((size_t)GATE_BLOCKS*E_*4);
    int*      cnt_part = (int*)     carve((size_t)GATE_BLOCKS*E_*4);
    int*      offs     = (int*)     carve(64);
    int*      counts2  = (int*)     carve(64);
    int*      brow     = (int*)     carve((size_t)NSLOT*4);
    float*    bgate    = (float*)   carve((size_t)NSLOT*4);
    int*      slot_of  = (int*)     carve((size_t)2*B_*4);
    float*    contrib  = (float*)   carve((size_t)NSLOT*2*4);
    unsigned* w1pk     = (unsigned*)carve((size_t)(E_*IN_*H_/2)*4);   // 1 MB
    unsigned* w2pk     = (unsigned*)carve((size_t)(E_*H_*MO_/2)*4);   // 512 KB

    pack_w1_kernel<<<(E_*IN_*H_/2 + 255)/256, 256, 0, stream>>>(W1, w1pk);
    pack_w2_kernel<<<(E_*H_*MO_/2 + 255)/256, 256, 0, stream>>>(W2, w2pk);
    init_kernel<<<(NSLOT + 255)/256, 256, 0, stream>>>(brow, bgate);
    gate_kernel<<<GATE_BLOCKS, 256, 0, stream>>>(x, wg, top_idx, gate_val, imp_part, cnt_part);
    reduce_kernel<<<1, 256, 0, stream>>>(imp_part, cnt_part, offs, counts2, out + 2*B_);
    scatter_kernel<<<B_/256, 256, 0, stream>>>(top_idx, gate_val, offs, counts2, brow, bgate, slot_of);
    expert_kernel<<<NSLOT/TILE, 256, SMEM_BYTES, stream>>>(x, w1pk, w2pk, b1, b2, Wout, offs, brow, bgate, contrib);
    combine_kernel<<<B_/256, 256, 0, stream>>>(contrib, slot_of, bout, out);
}
